// GCNClassifer_6786048328674
// MI455X (gfx1250) — matrix operand order repaired and checked by reading, not compiled, not run on hardware
//
#include <hip/hip_runtime.h>
#include <math.h>

// ---------------------------------------------------------------------------
// GCN classifier pipeline for gfx1250 (MI455X), bf16 WMMA, f32 accumulate.
// Dominant cost: adj (16384^2) streamed twice through WMMA GEMMs -> use
// bf16 copy of adj + GLOBAL_LOAD_ASYNC_TO_LDS_B128 for its LDS staging.
// ---------------------------------------------------------------------------

#define BM 128
#define BN 128
#define BK 32
#define BKA 40   // padded LDS stride for A tile (elements)
#define BKB 40   // padded LDS stride for B tile (elements)

// gfx1250 async global->LDS path (ASYNCcnt), guarded so either toolchain builds.
#if defined(__gfx1250__) && __has_builtin(__builtin_amdgcn_global_load_async_to_lds_b128)
#define HAVE_ASYNC_LDS 1
#else
#define HAVE_ASYNC_LDS 0
#endif

typedef __attribute__((ext_vector_type(16))) __bf16        v16bf;
typedef __attribute__((ext_vector_type(8)))  float          v8f;
typedef __attribute__((ext_vector_type(16))) unsigned short v16u;
typedef __attribute__((ext_vector_type(8)))  unsigned short v8u;

#if HAVE_ASYNC_LDS
// Builtin signature (from toolchain diagnostics): params are pointers to
// 16-byte int vectors in addrspace(1) / addrspace(3), then imm offset, imm cpol.
typedef int v4i_gcc __attribute__((vector_size(16)));
typedef __attribute__((address_space(1))) v4i_gcc* gbl_v4i;
typedef __attribute__((address_space(3))) v4i_gcc* lds_v4i;
#endif

union BFV { v16u u; v16bf b; };

__device__ __forceinline__ void wait_async_lds() {
#if HAVE_ASYNC_LDS
#if __has_builtin(__builtin_amdgcn_s_wait_asynccnt)
  __builtin_amdgcn_s_wait_asynccnt(0);
#else
  asm volatile("s_wait_asynccnt 0x0" ::: "memory");
#endif
#endif
}

// float -> bf16 (round-to-nearest-even), pure integer math (host-pass safe)
__device__ __forceinline__ unsigned short f2bf(float f) {
  union { float f; unsigned u; } c; c.f = f;
  unsigned u = c.u;
  unsigned r = u + 0x7FFFu + ((u >> 16) & 1u);
  return (unsigned short)(r >> 16);
}

// assemble a 16-element ushort vector from two 16B LDS chunks
__device__ __forceinline__ v16u ld16(const unsigned short* p0,
                                     const unsigned short* p1) {
  v8u lo = *(const v8u*)p0;
  v8u hi = *(const v8u*)p1;
  return __builtin_shufflevector(lo, hi, 0,1,2,3,4,5,6,7,8,9,10,11,12,13,14,15);
}

// ---------------------------------------------------------------------------
// Fused adjacency pass: deg = adj.sum(axis=1), optional bf16 copy of adj.
// ---------------------------------------------------------------------------
__global__ __launch_bounds__(256)
void adj_rowsum_convert(const float* __restrict__ adj,
                        unsigned short* __restrict__ adjb,
                        float* __restrict__ deg, int n) {
  __shared__ float red[256];
  const int row = blockIdx.x;
  const int tid = threadIdx.x;
  const float4* src = (const float4*)(adj + (size_t)row * n);
  const int n4 = n >> 2;
  float s = 0.0f;
  for (int i = tid; i < n4; i += 256) {
    float4 v = src[i];
    s += v.x + v.y + v.z + v.w;
    if (adjb) {
      union { unsigned short h[4]; uint2 u; } p;
      p.h[0] = f2bf(v.x); p.h[1] = f2bf(v.y);
      p.h[2] = f2bf(v.z); p.h[3] = f2bf(v.w);
      *(uint2*)(adjb + (size_t)row * n + (size_t)i * 4) = p.u;
    }
  }
  red[tid] = s;
  __syncthreads();
  for (int st = 128; st > 0; st >>= 1) {
    if (tid < st) red[tid] += red[tid + st];
    __syncthreads();
  }
  if (tid == 0) deg[row] = red[0];
}

__global__ void make_dis(const float* __restrict__ deg,
                         float* __restrict__ dis, int n) {
  int i = blockIdx.x * blockDim.x + threadIdx.x;
  if (i < n) dis[i] = rsqrtf(deg[i]);
}

// Fold BatchNorm (eval) + encoder bias into per-column affine:
//   y = (xw + eb - m) * g/sqrt(v+eps) + bb  ==  xw*alpha + beta
__global__ void bn_fold(const float* __restrict__ eb, const float* __restrict__ g,
                        const float* __restrict__ bb, const float* __restrict__ m,
                        const float* __restrict__ v, float* __restrict__ alpha,
                        float* __restrict__ beta, int n) {
  int i = blockIdx.x * blockDim.x + threadIdx.x;
  if (i < n) {
    float s = g[i] * rsqrtf(v[i] + 1e-5f);
    alpha[i] = s;
    beta[i]  = (eb[i] - m[i]) * s + bb[i];
  }
}

// ---------------------------------------------------------------------------
// Generic WMMA GEMM: C[M,N] = epilogue(A[M,K] @ B[K,N])
// Block = 256 threads = 8 wave32 waves (4x2); wave tile 32x64 = 2x4 WMMA
// tiles of v_wmma_f32_16x16x32_bf16. Double-buffered LDS; bf16 A tiles are
// staged with GLOBAL_LOAD_ASYNC_TO_LDS_B128 when the toolchain exposes it.
// ---------------------------------------------------------------------------
template <int ABF, int BBF>
__global__ __launch_bounds__(256)
void gemm_bf16_wmma(const void* __restrict__ Av, int lda,
                    const void* __restrict__ Bv, int ldb,
                    int M, int N, int K,
                    const float* __restrict__ rowScale,
                    const float* __restrict__ colScale,
                    const float* __restrict__ colBias,
                    int doRelu, int outBF16,
                    void* __restrict__ Cv, int ldc) {
  __shared__ unsigned short sA[2][BM][BKA];
  __shared__ unsigned short sB[2][BN][BKB];

  const int tid  = threadIdx.x;
  const int lane = tid & 31;
  const int wave = tid >> 5;      // 0..7
  const int wm   = wave & 3;      // 4 waves along M (32 rows each)
  const int wn   = wave >> 2;     // 2 waves along N (64 cols each)
  const int row0 = blockIdx.x * BM;
  const int col0 = blockIdx.y * BN;

  // staging maps (256 threads move a 128x32 A tile and a 32x128 B tile)
  const int arow  = tid >> 1;          // 0..127
  const int acol  = (tid & 1) * 16;    // 0 / 16
  const int bkrow = tid >> 3;          // 0..31
  const int bnseg = (tid & 7) * 16;    // 0..112

  const int KT = K / BK;
  const bool useAsyncA = (ABF != 0) && (HAVE_ASYNC_LDS != 0);

  uint4  aU[2]; float4 aF[4];
  uint4  bU[2]; float4 bF[4];

  // --- async A staging: global bf16 tile straight into LDS (ASYNCcnt) ---
  auto asyncA = [&](int buf, int kt) {
#if HAVE_ASYNC_LDS
    const unsigned short* g = (const unsigned short*)Av +
        (size_t)(row0 + arow) * lda + (size_t)kt * BK + acol;
    unsigned short* l = &sA[buf][arow][acol];
    __builtin_amdgcn_global_load_async_to_lds_b128(
        (gbl_v4i)(void*)g,       (lds_v4i)(void*)l,       0, 0);
    __builtin_amdgcn_global_load_async_to_lds_b128(
        (gbl_v4i)(void*)(g + 8), (lds_v4i)(void*)(l + 8), 0, 0);
#else
    (void)buf; (void)kt;
#endif
  };

  auto gloadA = [&](int kt) {
    if (ABF) {
      const unsigned short* g = (const unsigned short*)Av +
          (size_t)(row0 + arow) * lda + (size_t)kt * BK + acol;
      aU[0] = ((const uint4*)g)[0];
      aU[1] = ((const uint4*)g)[1];
    } else {
      const float* g = (const float*)Av +
          (size_t)(row0 + arow) * lda + (size_t)kt * BK + acol;
      aF[0] = ((const float4*)g)[0];
      aF[1] = ((const float4*)g)[1];
      aF[2] = ((const float4*)g)[2];
      aF[3] = ((const float4*)g)[3];
    }
  };
  auto gloadB = [&](int kt) {
    if (BBF) {
      const unsigned short* g = (const unsigned short*)Bv +
          (size_t)(kt * BK + bkrow) * ldb + col0 + bnseg;
      bU[0] = ((const uint4*)g)[0];
      bU[1] = ((const uint4*)g)[1];
    } else {
      const float* g = (const float*)Bv +
          (size_t)(kt * BK + bkrow) * ldb + col0 + bnseg;
      bF[0] = ((const float4*)g)[0];
      bF[1] = ((const float4*)g)[1];
      bF[2] = ((const float4*)g)[2];
      bF[3] = ((const float4*)g)[3];
    }
  };
  auto stageA = [&](int buf) {
    unsigned short* d = &sA[buf][arow][acol];
    if (ABF) {
      *(uint4*)d       = aU[0];
      *(uint4*)(d + 8) = aU[1];
    } else {
      const float* f = (const float*)&aF[0];
      unsigned short h[16];
      for (int i = 0; i < 16; ++i) h[i] = f2bf(f[i]);
      *(uint4*)d       = *(const uint4*)&h[0];
      *(uint4*)(d + 8) = *(const uint4*)&h[8];
    }
  };
  auto stageB = [&](int buf) {  // transpose into [n][k] so frag loads are contiguous
    if (BBF) {
      union { uint4 u[2]; unsigned short h[16]; } t;
      t.u[0] = bU[0]; t.u[1] = bU[1];
      for (int i = 0; i < 16; ++i) sB[buf][bnseg + i][bkrow] = t.h[i];
    } else {
      const float* f = (const float*)&bF[0];
      for (int i = 0; i < 16; ++i) sB[buf][bnseg + i][bkrow] = f2bf(f[i]);
    }
  };

  v8f acc[2][4];
  for (int i = 0; i < 2; ++i)
    for (int j = 0; j < 4; ++j)
      for (int v = 0; v < 8; ++v) acc[i][j][v] = 0.0f;

  // prologue: fill buffer 0
  if (useAsyncA) asyncA(0, 0); else gloadA(0);
  gloadB(0);
  if (!useAsyncA) stageA(0);
  stageB(0);
  if (useAsyncA) wait_async_lds();
  __syncthreads();

  const int lh = lane >> 4;   // 0/1
  const int lm = lane & 15;

  int buf = 0;
  for (int kt = 0; kt < KT; ++kt) {
    if (kt + 1 < KT) {
      if (useAsyncA) asyncA(buf ^ 1, kt + 1);  // issue before compute: latency hides
      else           gloadA(kt + 1);
      gloadB(kt + 1);
    }
    if (kt + 3 < KT) {
      // L2 prefetch of the streaming operand -> global_prefetch_b8
      if (ABF)
        __builtin_prefetch((const unsigned short*)Av +
            (size_t)(row0 + arow) * lda + (size_t)(kt + 3) * BK + acol, 0, 1);
      else
        __builtin_prefetch((const float*)Av +
            (size_t)(row0 + arow) * lda + (size_t)(kt + 3) * BK + acol, 0, 1);
    }

    // A fragments: 16-bit A 16x32 layout — lanes 0-15: K 0..7 / 16..23,
    // lanes 16-31: K 8..15 / 24..31 (ISA 7.12.2)
    BFV a[2], b[4];
    for (int i = 0; i < 2; ++i) {
      const unsigned short* base = &sA[buf][wm * 32 + i * 16 + lm][lh * 8];
      a[i].u = ld16(base, base + 16);
    }
    // B fragments: 32x16, lanes 0-15 hold K 0..7 / 16..23, lanes 16-31 hold K 8..15 / 24..31
    for (int j = 0; j < 4; ++j) {
      const unsigned short* base = &sB[buf][wn * 64 + j * 16 + lm][lh * 8];
      b[j].u = ld16(base, base + 16);
    }
    for (int i = 0; i < 2; ++i)
      for (int j = 0; j < 4; ++j)
        acc[i][j] = __builtin_amdgcn_wmma_f32_16x16x32_bf16(
            false, a[i].b, false, b[j].b, (short)0, acc[i][j], false, false);

    if (kt + 1 < KT) {
      if (!useAsyncA) stageA(buf ^ 1);
      stageB(buf ^ 1);
      if (useAsyncA) wait_async_lds();   // own waves' async loads landed
    }
    __syncthreads();
    buf ^= 1;
  }

  // Epilogue; C/D layout: VGPR v, lanes 0-15 -> M=v, lanes 16-31 -> M=v+8
  for (int i = 0; i < 2; ++i)
    for (int j = 0; j < 4; ++j) {
      const int gcol = col0 + wn * 64 + j * 16 + lm;
      const float cs = colScale ? colScale[gcol] : 1.0f;
      const float cb = colBias  ? colBias[gcol]  : 0.0f;
      for (int v = 0; v < 8; ++v) {
        const int grow = row0 + wm * 32 + i * 16 + lh * 8 + v;
        float x = acc[i][j][v];
        if (rowScale) x *= rowScale[grow];
        x = x * cs + cb;
        if (doRelu) x = fmaxf(x, 0.0f);
        if (outBF16)
          ((unsigned short*)Cv)[(size_t)grow * ldc + gcol] = f2bf(x);
        else
          ((float*)Cv)[(size_t)grow * ldc + gcol] = x;
      }
    }
}

// ---------------------------------------------------------------------------
// Tiny classifier head: out = clip(sigmoid(h4 @ cls_w + cls_b)), 16384x128x10
// ---------------------------------------------------------------------------
__global__ __launch_bounds__(256)
void classifier_kernel(const float* __restrict__ h, const float* __restrict__ w,
                       const float* __restrict__ b, float* __restrict__ out,
                       int M, int Kd) {
  int r = blockIdx.x * blockDim.x + threadIdx.x;
  if (r >= M) return;
  float acc[10];
  for (int c = 0; c < 10; ++c) acc[c] = b[c];
  const float* hr = h + (size_t)r * Kd;
  for (int k = 0; k < Kd; ++k) {
    float x = hr[k];
    const float* wr = w + (size_t)k * 10;
    for (int c = 0; c < 10; ++c) acc[c] = fmaf(x, wr[c], acc[c]);
  }
  float* o = out + (size_t)r * 10;
  for (int c = 0; c < 10; ++c) {
    float y = 1.0f / (1.0f + expf(-acc[c]));
    y = fminf(fmaxf(y, 1e-10f), 1.0f - 1e-10f);
    o[c] = y;
  }
}

// ---------------------------------------------------------------------------
extern "C" void kernel_launch(void* const* d_in, const int* in_sizes, int n_in,
                              void* d_out, int out_size, void* d_ws, size_t ws_size,
                              hipStream_t stream) {
  (void)in_sizes; (void)n_in; (void)out_size;
  const int N = 16384, F = 1024, H1 = 512, E = 256, H = 256, G = 128;

  const float* feature = (const float*)d_in[0];
  const float* adj     = (const float*)d_in[1];
  const float* enc_w1  = (const float*)d_in[2];
  const float* enc_b1  = (const float*)d_in[3];
  const float* bn1_g   = (const float*)d_in[4];
  const float* bn1_b   = (const float*)d_in[5];
  const float* bn1_m   = (const float*)d_in[6];
  const float* bn1_v   = (const float*)d_in[7];
  const float* enc_w2  = (const float*)d_in[8];
  const float* enc_b2  = (const float*)d_in[9];
  const float* bn2_g   = (const float*)d_in[10];
  const float* bn2_b   = (const float*)d_in[11];
  const float* bn2_m   = (const float*)d_in[12];
  const float* bn2_v   = (const float*)d_in[13];
  const float* gcn1_w  = (const float*)d_in[14];
  const float* gcn1_b  = (const float*)d_in[15];
  const float* gcn2_w  = (const float*)d_in[16];
  const float* gcn2_b  = (const float*)d_in[17];
  const float* cls_w   = (const float*)d_in[18];
  const float* cls_b   = (const float*)d_in[19];

  char* ws = (char*)d_ws;
  size_t off = 0;
  auto alloc = [&](size_t bytes) {
    size_t o = off;
    off = (off + bytes + 255) & ~(size_t)255;
    return o;
  };
  float* deg = (float*)(ws + alloc((size_t)N * 4));
  float* dis = (float*)(ws + alloc((size_t)N * 4));
  float* al1 = (float*)(ws + alloc((size_t)H1 * 4));
  float* be1 = (float*)(ws + alloc((size_t)H1 * 4));
  float* al2 = (float*)(ws + alloc((size_t)E * 4));
  float* be2 = (float*)(ws + alloc((size_t)E * 4));
  unsigned short* h1 = (unsigned short*)(ws + alloc((size_t)N * H1 * 2));
  unsigned short* h2 = (unsigned short*)(ws + alloc((size_t)N * E * 2));
  unsigned short* t1 = (unsigned short*)(ws + alloc((size_t)N * H * 2));
  unsigned short* h3 = (unsigned short*)(ws + alloc((size_t)N * H * 2));
  unsigned short* t2 = (unsigned short*)(ws + alloc((size_t)N * G * 2));
  float*          h4 = (float*)(ws + alloc((size_t)N * G * 4));

  // bf16 adjacency copy (512 MB) if workspace allows; else stream fp32 adj.
  size_t adjb_off   = off;
  size_t adjb_bytes = (size_t)N * N * 2;
  bool   adj_bf     = (adjb_off + adjb_bytes) <= ws_size;
  unsigned short* adjb = adj_bf ? (unsigned short*)(ws + adjb_off) : nullptr;

  // 1) fused deg + bf16 conversion of adj (single fp32 read of adj)
  adj_rowsum_convert<<<N, 256, 0, stream>>>(adj, adjb, deg, N);
  make_dis<<<(N + 255) / 256, 256, 0, stream>>>(deg, dis, N);
  bn_fold<<<(H1 + 255) / 256, 256, 0, stream>>>(enc_b1, bn1_g, bn1_b, bn1_m, bn1_v, al1, be1, H1);
  bn_fold<<<(E  + 255) / 256, 256, 0, stream>>>(enc_b2, bn2_g, bn2_b, bn2_m, bn2_v, al2, be2, E);

  // 2) h1 = relu(alpha1 * (feature @ W1) + beta1)
  {
    dim3 g(N / BM, H1 / BN);
    gemm_bf16_wmma<0, 0><<<g, 256, 0, stream>>>(
        feature, F, enc_w1, H1, N, H1, F, nullptr, al1, be1, 1, 1, h1, H1);
  }
  // 3) h2 = relu(alpha2 * (h1 @ W2) + beta2)
  {
    dim3 g(N / BM, E / BN);
    gemm_bf16_wmma<1, 0><<<g, 256, 0, stream>>>(
        h1, H1, enc_w2, E, N, E, H1, nullptr, al2, be2, 1, 1, h2, E);
  }
  // 4) t1 = dis .* (h2 @ gcn1_w)
  {
    dim3 g(N / BM, H / BN);
    gemm_bf16_wmma<1, 0><<<g, 256, 0, stream>>>(
        h2, E, gcn1_w, H, N, H, E, dis, nullptr, nullptr, 0, 1, t1, H);
  }
  // 5) h3 = relu(dis .* (adj @ t1) + gcn1_b)
  {
    dim3 g(N / BM, H / BN);
    if (adj_bf)
      gemm_bf16_wmma<1, 1><<<g, 256, 0, stream>>>(
          adjb, N, t1, H, N, H, N, dis, nullptr, gcn1_b, 1, 1, h3, H);
    else
      gemm_bf16_wmma<0, 1><<<g, 256, 0, stream>>>(
          adj, N, t1, H, N, H, N, dis, nullptr, gcn1_b, 1, 1, h3, H);
  }
  // 6) t2 = dis .* (h3 @ gcn2_w)
  {
    dim3 g(N / BM, G / BN);
    gemm_bf16_wmma<1, 0><<<g, 256, 0, stream>>>(
        h3, H, gcn2_w, G, N, G, H, dis, nullptr, nullptr, 0, 1, t2, G);
  }
  // 7) h4 = relu(dis .* (adj @ t2) + gcn2_b)   (fp32 out for the head)
  {
    dim3 g(N / BM, G / BN);
    if (adj_bf)
      gemm_bf16_wmma<1, 1><<<g, 256, 0, stream>>>(
          adjb, N, t2, G, N, G, N, dis, nullptr, gcn2_b, 1, 0, h4, G);
    else
      gemm_bf16_wmma<0, 1><<<g, 256, 0, stream>>>(
          adj, N, t2, G, N, G, N, dis, nullptr, gcn2_b, 1, 0, h4, G);
  }
  // 8) out = clip(sigmoid(h4 @ cls_w + cls_b))
  classifier_kernel<<<N / 256, 256, 0, stream>>>(h4, cls_w, cls_b,
                                                 (float*)d_out, N, G);
}
